// LGBClassifier_31353261261433
// MI455X (gfx1250) — matrix operand order repaired and checked by reading, not compiled, not run on hardware
//
#include <hip/hip_runtime.h>
#include <hip/hip_bf16.h>

#define TREE_DEPTH 5
#define N_NODES 31
#define N_LEAVES 32
#define F_DIM 128
#define EPS 1e-7f
#define POSPAD 20            /* floats per [pos] row in LDS (16 rows + pad) */

typedef __attribute__((ext_vector_type(16))) __bf16          v16bf;
typedef __attribute__((ext_vector_type(16))) unsigned short  v16us;
typedef __attribute__((ext_vector_type(8)))  float           v8f;

union BFrag { v16us u; v16bf v; };

__device__ __forceinline__ unsigned short f2bf_rne(float f) {
  unsigned u = __float_as_uint(f);
  return (unsigned short)((u + 0x7FFFu + ((u >> 16) & 1u)) >> 16);
}
__device__ __forceinline__ float bf2f(unsigned short h) {
  return __uint_as_float(((unsigned)h) << 16);
}

// ---------------------------------------------------------------------------
// Prepack kernel[T,F,N] (f32) -> hi/lo bf16 in per-lane B-fragment order (contiguous k, same as A).
// kp[ (((t*4+s)*2+nt)*2+part)*512 + lane*16 + j ]
//   column n = (lane&15) + nt*16 (n==31 zero-padded); K = (lane>>4)*16 + j
// ---------------------------------------------------------------------------
__global__ void pack_kernel_b(const float* __restrict__ kern,
                              unsigned short* __restrict__ kp, int T) {
  int idx = blockIdx.x * 256 + threadIdx.x;
  int total = T * 4 * 2 * 512;
  if (idx >= total) return;
  int pos  = idx & 511;
  int lane = pos >> 4;
  int j    = pos & 15;
  int nt   = (idx >> 9) & 1;
  int s    = (idx >> 10) & 3;
  int t    = idx >> 12;
  int n = (lane & 15) + nt * 16;
  int K = (lane >> 4) * 16 + j;
  int f = s * 32 + K;
  float v = (n < N_NODES) ? kern[((size_t)t * F_DIM + f) * N_NODES + n] : 0.0f;
  unsigned short hi = f2bf_rne(v);
  unsigned short lo = f2bf_rne(v - bf2f(hi));
  size_t base = ((size_t)(((t * 4 + s) * 2 + nt) * 2)) * 512 + pos;
  kp[base]       = hi;
  kp[base + 512] = lo;
}

// ---------------------------------------------------------------------------
// Fused GBM: hi/lo-split bf16 WMMA GEMM + sigmoid (rcp/exp) + per-leaf
// probability products via b128 LDS tiles + leaf-weighted reduction.
// One wave owns 16 rows of x; 8 waves per workgroup, disjoint LDS slices.
// Branch-free tree loop: dead column n==31 writes to LDS positions 62/63,
// which the leaf stage never reads (max path position = 61).
// ---------------------------------------------------------------------------
__global__ __launch_bounds__(256, 1) void gbm_fused(
    const float* __restrict__ x,    // [B,128]
    const float* __restrict__ bias, // [T,31]
    const float* __restrict__ leaf, // [T,32]
    const unsigned short* __restrict__ kp,
    float* __restrict__ out, int T) {
  __shared__ float a1s[8][64][POSPAD];   // [wave][node*2+dir][row(16)+pad]

  const int lane = threadIdx.x & 31;
  const int wave = threadIdx.x >> 5;
  const int m    = lane & 15;
  const int lhi  = lane >> 4;
  const int row_base = (blockIdx.x * 8 + wave) * 16;
  const float* xrow = x + (size_t)(row_base + m) * F_DIM;

  // --- A fragments (hi/lo bf16), built once, reused for all trees ---
  BFrag Ah[4], Al[4];
#pragma unroll
  for (int s = 0; s < 4; ++s) {
    const float* p0 = xrow + s * 32 + lhi * 16;
    float4 a = *(const float4*)(p0);
    float4 b = *(const float4*)(p0 + 4);
    float4 c = *(const float4*)(p0 + 8);
    float4 d = *(const float4*)(p0 + 12);
    float vals[16] = {a.x,a.y,a.z,a.w, b.x,b.y,b.z,b.w,
                      c.x,c.y,c.z,c.w, d.x,d.y,d.z,d.w};
#pragma unroll
    for (int j = 0; j < 16; ++j) {
      unsigned short h = f2bf_rne(vals[j]);
      Ah[s].u[j] = h;
      Al[s].u[j] = f2bf_rne(vals[j] - bf2f(h));
    }
  }

  // --- per-lane (== leaf) path offsets, hoisted out of the tree loop ---
  int off[TREE_DEPTH];
  {
    int node = 0;
#pragma unroll
    for (int d = 0; d < TREE_DEPTH; ++d) {
      int bit = (lane >> (4 - d)) & 1;
      off[d] = (node * 2 + bit) * POSPAD;
      node = 2 * node + 1 + bit;
    }
  }

  float accm[16];
#pragma unroll
  for (int i = 0; i < 16; ++i) accm[i] = 0.f;

  float* base = &a1s[wave][0][0];
  const int n1  = 16 + m;
  const int n1c = (n1 < N_NODES) ? n1 : (N_NODES - 1);  // clamp: value unused for n1==31

  for (int t = 0; t < T; ++t) {
    float leafv = leaf[t * N_LEAVES + lane];
    float b0 = bias[t * N_NODES + m];
    float b1 = bias[t * N_NODES + n1c];   // unconditional; dead col never read

    // ---------------- GEMM: h[16 x 32] in two accumulator fragments --------
    v8f c0 = {}; v8f c1 = {};
    const unsigned short* kt = kp + (size_t)t * (4 * 2 * 2 * 512);
#pragma unroll
    for (int s = 0; s < 4; ++s) {
      const unsigned short* ks = kt + s * 2048 + lane * 16;
      v16bf bh0 = *(const v16bf*)(ks);
      v16bf bl0 = *(const v16bf*)(ks + 512);
      v16bf bh1 = *(const v16bf*)(ks + 1024);
      v16bf bl1 = *(const v16bf*)(ks + 1536);
      c0 = __builtin_amdgcn_wmma_f32_16x16x32_bf16(false, Ah[s].v, false, bh0, (short)0, c0, false, false);
      c1 = __builtin_amdgcn_wmma_f32_16x16x32_bf16(false, Ah[s].v, false, bh1, (short)0, c1, false, false);
      c0 = __builtin_amdgcn_wmma_f32_16x16x32_bf16(false, Al[s].v, false, bh0, (short)0, c0, false, false);
      c1 = __builtin_amdgcn_wmma_f32_16x16x32_bf16(false, Al[s].v, false, bh1, (short)0, c1, false, false);
      c0 = __builtin_amdgcn_wmma_f32_16x16x32_bf16(false, Ah[s].v, false, bl0, (short)0, c0, false, false);
      c1 = __builtin_amdgcn_wmma_f32_16x16x32_bf16(false, Ah[s].v, false, bl1, (short)0, c1, false, false);
    }

    // ------------- sigmoid pairs, packed per-position into LDS -------------
    // D-frag element r: row = r + 8*lhi, col n = m (c0) / 16+m (c1)
    float s0[8], s1[8], s2[8], s3[8];
#pragma unroll
    for (int r = 0; r < 8; ++r) {
      {
        float h  = c0[r] + b0;
        float zc = fminf(fmaxf(h * 100.f, -32.f), 32.f);
        float e  = __expf(-zc);
        float sg = __builtin_amdgcn_rcpf(1.f + e);   // sigmoid(zc)
        s0[r] = fmaxf(sg,     EPS);                  // dir 0 (h)
        s1[r] = fmaxf(e * sg, EPS);                  // dir 1 (-h)
      }
      {
        float h  = c1[r] + b1;
        float zc = fminf(fmaxf(h * 100.f, -32.f), 32.f);
        float e  = __expf(-zc);
        float sg = __builtin_amdgcn_rcpf(1.f + e);
        s2[r] = fmaxf(sg,     EPS);
        s3[r] = fmaxf(e * sg, EPS);
      }
    }
    {
      float* p = base + (2 * m) * POSPAD + lhi * 8;
      *(float4*)(p)     = float4{s0[0], s0[1], s0[2], s0[3]};
      *(float4*)(p + 4) = float4{s0[4], s0[5], s0[6], s0[7]};
      p += POSPAD;
      *(float4*)(p)     = float4{s1[0], s1[1], s1[2], s1[3]};
      *(float4*)(p + 4) = float4{s1[4], s1[5], s1[6], s1[7]};
    }
    {
      // n1==31 lanes write positions 62/63: never read, harmless.
      float* p = base + (2 * n1) * POSPAD + lhi * 8;
      *(float4*)(p)     = float4{s2[0], s2[1], s2[2], s2[3]};
      *(float4*)(p + 4) = float4{s2[4], s2[5], s2[6], s2[7]};
      p += POSPAD;
      *(float4*)(p)     = float4{s3[0], s3[1], s3[2], s3[3]};
      *(float4*)(p + 4) = float4{s3[4], s3[5], s3[6], s3[7]};
    }
    asm volatile("s_wait_dscnt 0" ::: "memory");  // same-wave LDS fence

    // ------- lane == leaf: prob = product of 5 sigmoids, 4 rows/load -------
#pragma unroll
    for (int blk = 0; blk < 4; ++blk) {
      const float* bp = base + blk * 4;
      float4 q0 = *(const float4*)(bp + off[0]);
      float4 q1 = *(const float4*)(bp + off[1]);
      float4 q2 = *(const float4*)(bp + off[2]);
      float4 q3 = *(const float4*)(bp + off[3]);
      float4 q4 = *(const float4*)(bp + off[4]);
      float a0[4] = {q0.x, q0.y, q0.z, q0.w};
      float a1[4] = {q1.x, q1.y, q1.z, q1.w};
      float a2[4] = {q2.x, q2.y, q2.z, q2.w};
      float a3[4] = {q3.x, q3.y, q3.z, q3.w};
      float a4[4] = {q4.x, q4.y, q4.z, q4.w};
#pragma unroll
      for (int r2 = 0; r2 < 4; ++r2) {
        float prob = a0[r2] * a1[r2] * a2[r2] * a3[r2] * a4[r2];
        accm[blk * 4 + r2] = fmaf(prob, leafv, accm[blk * 4 + r2]);
      }
    }
    asm volatile("" ::: "memory");
  }

  // cross-lane (leaf) reduction for the 16 rows
  float res = 0.f;
#pragma unroll
  for (int mm = 0; mm < 16; ++mm) {
    float v = accm[mm];
#pragma unroll
    for (int o = 16; o > 0; o >>= 1) v += __shfl_xor(v, o, 32);
    if (lane == mm) res = v;
  }
  if (lane < 16) out[row_base + lane] = res;
}

// ---------------------------------------------------------------------------
extern "C" void kernel_launch(void* const* d_in, const int* in_sizes, int n_in,
                              void* d_out, int out_size, void* d_ws, size_t ws_size,
                              hipStream_t stream) {
  const float* x    = (const float*)d_in[0];
  const float* kern = (const float*)d_in[1];
  const float* bias = (const float*)d_in[2];
  const float* leaf = (const float*)d_in[3];
  // d_in[4] = route: fixed perfect-tree topology, paths computed inline.

  int B = in_sizes[0] / F_DIM;
  int T = in_sizes[3] / N_LEAVES;

  unsigned short* kp = (unsigned short*)d_ws;   // T*4*2*2*512 ushorts (~1.6 MB)
  int total = T * 4 * 2 * 512;
  pack_kernel_b<<<(total + 255) / 256, 256, 0, stream>>>(kern, kp, T);
  gbm_fused<<<B / 128, 256, 0, stream>>>(x, bias, leaf, kp, (float*)d_out, T);
}
